// GraphConvolution_82944408420470
// MI455X (gfx1250) — matrix operand order repaired and checked by reading, not compiled, not run on hardware
//
#include <hip/hip_runtime.h>

// GC-MC graph convolution for MI455X (gfx1250, wave32).
// v4: relation weights staged per-k in LDS (bf16, bank-conflict-free padded
// layout), shared by all 8 waves; v_wmma_f32_16x16x32_bf16 fed by ds_load.
// The per-row relation select is a masked-FMA accumulation (v_cndmask +
// v_fmac) instead of a ternary, so the compiler cannot turn it into
// exec-mask branch diamonds that break the WMMA pipeline (seen in v3).

typedef __attribute__((ext_vector_type(16))) __bf16 v16bf;
typedef __attribute__((ext_vector_type(8)))  __bf16 v8bf;
typedef __attribute__((ext_vector_type(8)))  float  v8f;

#define FIN   128
#define HDIM  64
#define ODIM  32
#define KREL  5
#define WPB   8            // waves per block
#define TSTRIDE 72         // transpose-tile row stride (bf16): 64 + 8 pad
#define WSTRIDE 136        // staged-weight row stride (bf16): 128 + 8 pad
                           // -> 68 dwords/row: lane lo starts at bank 4*lo,
                           //    16-byte B-fragment reads cover all 64 banks

#define NWU (KREL * HDIM * FIN)   // 40960 elements (Wu), same for Wv
#define NWL (ODIM * HDIM)         // 2048 elements  (Wl)
#define WS_WU 0
#define WS_WV NWU
#define WS_WL (2 * NWU)
#define WS_TOTAL (2 * NWU + NWL)  // 83968 bf16 = 164 KB of d_ws

// ---------------- prologue: fp32 weights -> bf16 workspace ----------------
__global__ void cvt_weights_kernel(const float* __restrict__ Wu,
                                   const float* __restrict__ Wv,
                                   const float* __restrict__ Wl,
                                   __bf16* __restrict__ ws) {
  const int i = blockIdx.x * blockDim.x + threadIdx.x;
  if (i < NWU)           ws[i] = (__bf16)Wu[i];
  else if (i < 2 * NWU)  ws[i] = (__bf16)Wv[i - NWU];
  else if (i < WS_TOTAL) ws[i] = (__bf16)Wl[i - 2 * NWU];
}

// ---------------- fragment helpers ----------------

// B-fragment for v_wmma_f32_16x16x32_bf16: lane L (<16) holds column n0+L
// with K = f0..f0+7, f0+16..f0+23 (two 16-byte loads); lane L+16
// holds K = f0+8..f0+15, f0+24..f0+31. Address space inferred after inlining
// (ds_load_b128 for LDS, global_load_b128 for global).
__device__ __forceinline__ v16bf ldB(const __bf16* p) {
  v16bf o;
  v8bf a = *(const v8bf*)p;
  v8bf b = *(const v8bf*)(p + 16);
#pragma unroll
  for (int i = 0; i < 8; ++i) { o[i] = a[i]; o[8 + i] = b[i]; }
  return o;
}

// A-fragment (16x32 bf16, row M = lane%16) from fp32 activations:
// elements 0..7 <- p[0..7], 8..15 <- p[16..23] (caller folds the +8 K-offset
// for lanes >= 16). fp32->bf16 conversion here is inherent (HBM stream).
__device__ __forceinline__ v16bf cvtA(const float* __restrict__ p) {
  v16bf o;
  const float4* q0 = (const float4*)p;
  const float4* q1 = (const float4*)(p + 16);
#pragma unroll
  for (int i = 0; i < 2; ++i) {
    float4 f = q0[i];
    o[4 * i + 0] = (__bf16)f.x;
    o[4 * i + 1] = (__bf16)f.y;
    o[4 * i + 2] = (__bf16)f.z;
    o[4 * i + 3] = (__bf16)f.w;
  }
#pragma unroll
  for (int i = 0; i < 2; ++i) {
    float4 f = q1[i];
    o[8 + 4 * i + 0] = (__bf16)f.x;
    o[8 + 4 * i + 1] = (__bf16)f.y;
    o[8 + 4 * i + 2] = (__bf16)f.z;
    o[8 + 4 * i + 3] = (__bf16)f.w;
  }
  return o;
}

__device__ __forceinline__ v8f wmma_bf16(v16bf a, v16bf b, v8f c) {
  return __builtin_amdgcn_wmma_f32_16x16x32_bf16(
      false, a, false, b, (short)0, c, false, false);
}

// bias + c-scale + ReLU + 64->32 GEMM + ReLU + store for one path.
__device__ __forceinline__ void epilogue(const v8f* sel,
                                         const float*  __restrict__ brel,
                                         const __bf16* __restrict__ WlBf,
                                         const float*  __restrict__ bl,
                                         const int* rv, const float* cv,
                                         float* __restrict__ out,
                                         int row0, int N,
                                         __bf16* __restrict__ t,  // per-wave
                                         int lo, int hi, int hiOff) {
  // D layout: (vgpr j, lane) -> row M = j + hi*8, col = lo.
#pragma unroll
  for (int nt = 0; nt < 4; ++nt) {
    const int h0 = nt * 16;
#pragma unroll
    for (int j = 0; j < 8; ++j) {
      float v = cv[j] * (sel[nt][j] + brel[rv[j] * HDIM + h0 + lo]);
      v = v > 0.0f ? v : 0.0f;
      t[(j + hi * 8) * TSTRIDE + h0 + lo] = (__bf16)v;
    }
  }
  // Same-wave LDS ops are in-order; tile is wave-private => no barrier.

  v16bf a2[2];
#pragma unroll
  for (int ks = 0; ks < 2; ++ks) {
    const __bf16* p = t + lo * TSTRIDE + ks * 32 + hiOff;
#pragma unroll
    for (int e = 0; e < 8; ++e) {
      a2[ks][e]     = p[e];
      a2[ks][8 + e] = p[16 + e];
    }
  }

#pragma unroll
  for (int ot = 0; ot < 2; ++ot) {
    const int o0 = ot * 16;
    v8f acc;
#pragma unroll
    for (int j = 0; j < 8; ++j) acc[j] = 0.0f;

    const __bf16* wl = WlBf + (o0 + lo) * HDIM + hi * 8;
#pragma unroll
    for (int ks = 0; ks < 2; ++ks)
      acc = wmma_bf16(a2[ks], ldB(wl + ks * 32), acc);

    const float bb = bl[o0 + lo];
#pragma unroll
    for (int j = 0; j < 8; ++j) {
      const int row = row0 + j + hi * 8;
      if (row < N) {
        float v = acc[j] + bb;
        out[(size_t)row * ODIM + o0 + lo] = v > 0.0f ? v : 0.0f;
      }
    }
  }
}

// ---------------- main kernel: one wave32 per 16-row tile ----------------
__global__ __launch_bounds__(32 * WPB)
void gcmc_kernel(const float* __restrict__ user,
                 const float* __restrict__ item,
                 const int*   __restrict__ r,
                 const float* __restrict__ c,
                 const __bf16* __restrict__ wbf,   // [Wu | Wv | Wl] bf16
                 const float* __restrict__ bu,
                 const float* __restrict__ bv,
                 const float* __restrict__ bl,
                 float* __restrict__ out_u,
                 float* __restrict__ out_v,
                 int N, int ntiles) {
  __shared__ __align__(16) __bf16 sWu[HDIM * WSTRIDE];        // 17 KB
  __shared__ __align__(16) __bf16 sWv[HDIM * WSTRIDE];        // 17 KB
  __shared__ __align__(16) __bf16 sT[WPB][16 * TSTRIDE];      // 18 KB

  const int tid   = threadIdx.x;
  const int wave  = tid >> 5;                 // wave-uniform
  const int lane  = tid & 31;
  const int lo    = lane & 15;
  const int hi    = lane >> 4;
  const int hiOff = hi * 8;
  const int tile  = blockIdx.x * WPB + wave;  // tail waves: rows clamp, no store
  const int row0  = tile * 16;

  // per-row relation index / coefficient (D-row M = j + hi*8)
  int   rv[8];
  float cv[8];
#pragma unroll
  for (int j = 0; j < 8; ++j) {
    int row = row0 + j + hi * 8;
    row = row < N ? row : N - 1;
    rv[j] = r[row];
    cv[j] = c[row];
  }

  // A fragments for both paths (row M = lo), streamed once from HBM
  int arow = row0 + lo;
  arow = arow < N ? arow : N - 1;
  const float* irow = item + (size_t)arow * FIN;
  const float* urow = user + (size_t)arow * FIN;
  v16bf au[4], av[4];
#pragma unroll
  for (int ks = 0; ks < 4; ++ks) {
    au[ks] = cvtA(irow + ks * 32 + hiOff);
    av[ks] = cvtA(urow + ks * 32 + hiOff);
  }

  v8f selu[4], selv[4];
#pragma unroll
  for (int nt = 0; nt < 4; ++nt)
#pragma unroll
    for (int j = 0; j < 8; ++j) { selu[nt][j] = 0.0f; selv[nt][j] = 0.0f; }

  // ---- Stage 1: loop relations; weights staged in LDS per k ----
  for (int k = 0; k < KREL; ++k) {
    __syncthreads();   // previous k's compute done before overwrite
    {
      const __bf16* gu = wbf + WS_WU + (size_t)k * HDIM * FIN;
      const __bf16* gv = wbf + WS_WV + (size_t)k * HDIM * FIN;
      // 64 rows x 128 bf16 = 1024 16B-chunks; 4 chunks per thread
#pragma unroll
      for (int i = 0; i < 4; ++i) {
        const int chunk = tid + i * 256;
        const int row = chunk >> 4, col = chunk & 15;
        v8bf tu = *(const v8bf*)(gu + row * FIN + col * 8);
        v8bf tv = *(const v8bf*)(gv + row * FIN + col * 8);
        *(v8bf*)(sWu + row * WSTRIDE + col * 8) = tu;
        *(v8bf*)(sWv + row * WSTRIDE + col * 8) = tv;
      }
    }
    __syncthreads();

    // Branch-free relation mask: exactly one k matches each row, so the
    // running "selected" accumulator is sel += m * acc (v_cndmask + v_fmac;
    // exact, since m is 0.0 or 1.0).
    float mk[8];
#pragma unroll
    for (int j = 0; j < 8; ++j) mk[j] = (rv[j] == k) ? 1.0f : 0.0f;

#pragma unroll
    for (int nt = 0; nt < 4; ++nt) {
      const int h0 = nt * 16;
      const __bf16* pu = sWu + (h0 + lo) * WSTRIDE + hi * 8;
      const __bf16* pv = sWv + (h0 + lo) * WSTRIDE + hi * 8;
      v8f accu, accv;
#pragma unroll
      for (int j = 0; j < 8; ++j) { accu[j] = 0.0f; accv[j] = 0.0f; }
#pragma unroll
      for (int ks = 0; ks < 4; ++ks) {
        accu = wmma_bf16(au[ks], ldB(pu + ks * 32), accu);
        accv = wmma_bf16(av[ks], ldB(pv + ks * 32), accv);
      }
#pragma unroll
      for (int j = 0; j < 8; ++j) {
        selu[nt][j] = __builtin_fmaf(mk[j], accu[j], selu[nt][j]);
        selv[nt][j] = __builtin_fmaf(mk[j], accv[j], selv[nt][j]);
      }
    }
  }

  // ---- Stage 2 + stores (weight LDS no longer touched; sT is per-wave) ----
  const __bf16* WlBf = wbf + WS_WL;
  epilogue(selu, bu, WlBf, bl, rv, cv, out_u, row0, N, sT[wave], lo, hi, hiOff);
  epilogue(selv, bv, WlBf, bl, rv, cv, out_v, row0, N, sT[wave], lo, hi, hiOff);
}

extern "C" void kernel_launch(void* const* d_in, const int* in_sizes, int n_in,
                              void* d_out, int out_size, void* d_ws, size_t ws_size,
                              hipStream_t stream) {
  const float* user = (const float*)d_in[0];
  const float* item = (const float*)d_in[1];
  const int*   r    = (const int*)  d_in[2];
  const float* c    = (const float*)d_in[3];
  const float* Wu   = (const float*)d_in[4];
  const float* bu   = (const float*)d_in[5];
  const float* Wv   = (const float*)d_in[6];
  const float* bv   = (const float*)d_in[7];
  const float* Wl   = (const float*)d_in[8];
  const float* bl   = (const float*)d_in[9];
  float*  out = (float*)d_out;
  __bf16* wbf = (__bf16*)d_ws;        // 164 KB scratch for bf16 weights

  const int N      = in_sizes[0] / FIN;
  const int ntiles = (N + 15) / 16;
  const int blocks = (ntiles + WPB - 1) / WPB;

  // 1) convert weights fp32 -> bf16 once (stream-ordered before main kernel)
  cvt_weights_kernel<<<(WS_TOTAL + 255) / 256, 256, 0, stream>>>(Wu, Wv, Wl, wbf);

  // 2) fused graph-conv kernel
  gcmc_kernel<<<blocks, 32 * WPB, 0, stream>>>(
      user, item, r, c, wbf, bu, bv, bl,
      out, out + (size_t)N * ODIM, N, ntiles);
}
